// PPFEmbeddingSin_67448166416445
// MI455X (gfx1250) — matrix operand order repaired and checked by reading, not compiled, not run on hardware
//
#include <hip/hip_runtime.h>
#include <hip/hip_bf16.h>

// ---------------------------------------------------------------------------
// PPFEmbeddingSin for MI455X (gfx1250, wave32, WMMA)
//
// Outputs (concatenated in d_out, fp32):
//   glo : (512, 512, 256)   = (E_d + E_a) @ W_d^T + 2*b_d
//   loc : (512, 256)        = max_k ( A_patch @ W_patch^T + b_patch )
//
// WMMA plan: V_WMMA_F32_16X16X32_F16.  A = on-the-fly sinusoidal embedding
// (f16, built from v_sin/v_cos), B = pre-swizzled f16 weight fragments staged
// into LDS with GLOBAL_LOAD_ASYNC_TO_LDS_B128 (ASYNCcnt), C/D = f32.
// ---------------------------------------------------------------------------

typedef __attribute__((ext_vector_type(16))) _Float16 v16h;
typedef __attribute__((ext_vector_type(8)))  float    v8f;

#define M_NODES   512
#define N_POINTS  16384
#define D_MODEL   256
#define KNN       128
#define SCALE_2PI 6.283185307179586f
// omega_t for d=256: exp2(-log2(10000)/128 * t),  t = 0..127
#define OMEGA_C_FULL  (-0.103810252965f)
// omega_t for dq=64: exp2(-log2(10000)/32  * t),  t = 0..31
#define OMEGA_C_QUART (-0.415241011861f)

#define FRAG_BYTES (16 * 8 * 32 * 8 * 4)   // 128 KB of f16 fragments per weight

// ---------------------------------------------------------------------------
// B-fragment layout used here for V_WMMA_F32_16X16X32_F16 (wave32), contiguous k as in A:
//   B is 32x16 (K x N).  lanes 0-15 hold K=0..15 (VGPR v: K=2v,2v+1 packed),
//   lanes 16-31 hold K=16..31.  lane&15 = N column.
// Fragment storage: frag[((nt*8 + kt)*32 + lane)*8 + v] = half2 {W[c][k],W[c][k+1]}
// with c = nt*16 + (lane&15), k = kt*32 + (lane>>4)*16 + 2v.
// ---------------------------------------------------------------------------
__global__ __launch_bounds__(256) void prep_wfrag_kernel(
    const float* __restrict__ W, _Float16* __restrict__ frag)
{
    int gid = blockIdx.x * 256 + threadIdx.x;        // 0 .. 32767 (dwords)
    if (gid >= 16 * 8 * 32 * 8) return;
    int v    = gid & 7;
    int lane = (gid >> 3) & 31;
    int kt   = (gid >> 8) & 7;
    int nt   = (gid >> 11) & 15;
    int c  = nt * 16 + (lane & 15);
    int k0 = kt * 32 + (lane >> 4) * 16 + 2 * v;
    frag[gid * 2 + 0] = (_Float16)W[c * D_MODEL + k0 + 0];
    frag[gid * 2 + 1] = (_Float16)W[c * D_MODEL + k0 + 1];
}

// ---------------------------------------------------------------------------
// Cooperative async copy of the 128 KB fragment image into LDS.
// GVS mode: vdst = LDS byte offset, vaddr = per-lane byte offset, saddr = base.
// Tracked by ASYNCcnt; completed with s_wait_asynccnt + workgroup barrier.
// ---------------------------------------------------------------------------
__device__ __forceinline__ void stage_frags_to_lds(
    const _Float16* __restrict__ gsrc, v16h* smem, int tid, int nthreads)
{
    unsigned lds_base = (unsigned)(uintptr_t)(void*)smem;   // low 32b = LDS offset
    unsigned long long gbase = (unsigned long long)(uintptr_t)gsrc;
    const int chunks = FRAG_BYTES / 16 / nthreads;          // b128 per thread
    for (int it = 0; it < chunks; ++it) {
        unsigned off = (unsigned)(it * nthreads + tid) * 16u;
        unsigned dst = lds_base + off;
        asm volatile("global_load_async_to_lds_b128 %0, %1, %2"
                     :: "v"(dst), "v"(off), "s"(gbase) : "memory");
    }
    asm volatile("s_wait_asynccnt 0x0" ::: "memory");
    __syncthreads();
}

// ---------------------------------------------------------------------------
// d2[m][n] = |node_m|^2 + |p_n|^2 - 2 node_m . p_n   (mirrors reference)
// ---------------------------------------------------------------------------
__global__ __launch_bounds__(256) void d2_kernel(
    const float* __restrict__ nodes, const float* __restrict__ points,
    float* __restrict__ d2)
{
    int n = blockIdx.x * 256 + threadIdx.x;          // point index
    int m = blockIdx.y;                              // node index
    float px = points[3 * n + 0], py = points[3 * n + 1], pz = points[3 * n + 2];
    float qx = nodes[3 * m + 0],  qy = nodes[3 * m + 1],  qz = nodes[3 * m + 2];
    float sqp = px * px + py * py + pz * pz;
    float sqn = qx * qx + qy * qy + qz * qz;
    float dot = px * qx + py * qy + pz * qz;
    d2[(long)m * N_POINTS + n] = sqn + sqp - 2.0f * dot;
}

// ---------------------------------------------------------------------------
// Per-node top-K (K=128, ascending, ties -> lower index), no mutation:
// each of 128 rounds selects the minimum composite key (d2, idx) strictly
// greater than the previously selected key.  64 d2 values per thread stay in
// VGPRs; cross-thread reduction through LDS.
// ---------------------------------------------------------------------------
__global__ __launch_bounds__(256) void knn_select_kernel(
    const float* __restrict__ d2, int* __restrict__ knn_idx)
{
    __shared__ float sd[256];
    __shared__ int   si[256];
    __shared__ float bc_d;
    __shared__ int   bc_i;

    int m = blockIdx.x, tid = threadIdx.x;
    const float* row = d2 + (long)m * N_POINTS;

    float v[64];
#pragma unroll
    for (int s = 0; s < 64; ++s) v[s] = row[s * 256 + tid];

    float lastd = -3.402823466e38f;
    int   lasti = -1;

    for (int sel = 0; sel < KNN; ++sel) {
        float bd = 3.402823466e38f;
        int   bi = 0x7fffffff;
#pragma unroll
        for (int s = 0; s < 64; ++s) {
            int   idx = s * 256 + tid;
            float dv  = v[s];
            bool gt = (dv > lastd) || (dv == lastd && idx > lasti);
            bool lt = (dv < bd)    || (dv == bd    && idx < bi);
            if (gt && lt) { bd = dv; bi = idx; }
        }
        sd[tid] = bd; si[tid] = bi;
        __syncthreads();
        for (int off = 128; off > 0; off >>= 1) {
            if (tid < off) {
                float od = sd[tid + off]; int oi = si[tid + off];
                if ((od < sd[tid]) || (od == sd[tid] && oi < si[tid])) {
                    sd[tid] = od; si[tid] = oi;
                }
            }
            __syncthreads();
        }
        if (tid == 0) {
            knn_idx[m * KNN + sel] = si[0];
            bc_d = sd[0]; bc_i = si[0];
        }
        __syncthreads();
        lastd = bc_d; lasti = bc_i;
    }
}

// ---------------------------------------------------------------------------
// glo kernel: block = 256 threads = 8 waves.  W_d fragments (128 KB) are async
// copied into LDS once per workgroup; each wave owns 16 rows of the flattened
// (M*M) row space and sweeps all 256 output columns, feeding WMMA from
// ds_load_b128.  A fragments are synthesized per wave from the two row scalars
// (distance, angle): K even/odd pairs == sin/cos of the same omega_t.
// ---------------------------------------------------------------------------
__global__ __launch_bounds__(256) void glo_kernel(
    const float* __restrict__ nodes, const float* __restrict__ nodes_normals,
    const _Float16* __restrict__ wd_frag, const float* __restrict__ b_d,
    float* __restrict__ glo)
{
    extern __shared__ v16h smem_frag[];                 // 128 KB dynamic LDS

    const int tid  = threadIdx.x;
    const int wave = tid >> 5;
    const int lane = tid & 31;
    const int lrow = lane & 15;
    const int hi   = lane >> 4;

    stage_frags_to_lds(wd_frag, smem_frag, tid, 256);

    const long rbase = (long)blockIdx.x * 128 + wave * 16;
    const long row   = rbase + lrow;            // this lane's matrix row
    const int  i = (int)(row >> 9);
    const int  j = (int)(row & (M_NODES - 1));

    // ---- per-row scalars: d_idx and a_ij_idx --------------------------------
    float dx = nodes[3 * i + 0] - nodes[3 * j + 0];
    float dy = nodes[3 * i + 1] - nodes[3 * j + 1];
    float dz = nodes[3 * i + 2] - nodes[3 * j + 2];
    float xd = sqrtf(dx * dx + dy * dy + dz * dz) * SCALE_2PI;
    float nx = nodes_normals[3 * i + 0];
    float ny = nodes_normals[3 * i + 1];
    float nz = nodes_normals[3 * i + 2];
    float cx = dy * nz - dz * ny;
    float cy = dz * nx - dx * nz;
    float cz = dx * ny - dy * nx;
    float xa = atan2f(sqrtf(cx * cx + cy * cy + cz * cz),
                      dx * nx + dy * ny + dz * nz);

    // ---- build all 8 A fragments (K = 0..255) -------------------------------
    v16h A[8];
#pragma unroll
    for (int kt = 0; kt < 8; ++kt) {
        v16h a;
#pragma unroll
        for (int v = 0; v < 8; ++v) {
            int t = kt * 16 + v + hi * 8;   // omega index
            float om = exp2f(OMEGA_C_FULL * (float)t);
            float ad = om * xd, aa = om * xa;
            float s = __sinf(ad) + __sinf(aa);
            float c = __cosf(ad) + __cosf(aa);
            a[2 * v + 0] = (_Float16)s;
            a[2 * v + 1] = (_Float16)c;
        }
        A[kt] = a;
    }

    // ---- sweep 16 column tiles, 8 K-steps each, B from LDS ------------------
    for (int nt = 0; nt < 16; ++nt) {
        v8f acc;
#pragma unroll
        for (int v = 0; v < 8; ++v) acc[v] = 0.0f;
#pragma unroll
        for (int kt = 0; kt < 8; ++kt) {
            v16h b = smem_frag[(nt * 8 + kt) * 32 + lane];
            acc = __builtin_amdgcn_wmma_f32_16x16x32_f16(
                false, A[kt], false, b, (short)0, acc, false, false);
        }
        int col = nt * 16 + lrow;
        float bias = 2.0f * b_d[col];
#pragma unroll
        for (int v = 0; v < 8; ++v) {
            long r = rbase + v + 8 * hi;
            glo[r * D_MODEL + col] = acc[v] + bias;
        }
    }
}

// ---------------------------------------------------------------------------
// loc kernel: one block (4 waves) per node, W_patch fragments staged into LDS.
// Wave w owns columns [w*64,w*64+64).  For each of 8 row tiles (16 neighbors):
// build the 4-scalar PPF features per row, synthesize quarter-embeddings as A
// fragments, WMMA against W_patch fragments from LDS, then fold the K-dim max
// into register fmax + lane-xor-16 shuffle.
// ---------------------------------------------------------------------------
__global__ __launch_bounds__(128) void loc_kernel(
    const float* __restrict__ points, const float* __restrict__ points_normals,
    const float* __restrict__ nodes,  const float* __restrict__ nodes_normals,
    const int* __restrict__ knn_idx,  const _Float16* __restrict__ wp_frag,
    const float* __restrict__ b_patch, float* __restrict__ loc)
{
    extern __shared__ v16h smem_frag[];                 // 128 KB dynamic LDS

    const int m    = blockIdx.x;
    const int tid  = threadIdx.x;
    const int wave = tid >> 5;
    const int lane = tid & 31;
    const int lrow = lane & 15;
    const int hi   = lane >> 4;

    stage_frags_to_lds(wp_frag, smem_frag, tid, 128);

    const float qx = nodes[3 * m + 0], qy = nodes[3 * m + 1], qz = nodes[3 * m + 2];
    const float ax = nodes_normals[3 * m + 0];
    const float ay = nodes_normals[3 * m + 1];
    const float az = nodes_normals[3 * m + 2];

    v8f vmax[4];
#pragma unroll
    for (int nt = 0; nt < 4; ++nt)
#pragma unroll
        for (int v = 0; v < 8; ++v) vmax[nt][v] = -3.402823466e38f;

    for (int rt = 0; rt < 8; ++rt) {
        // ---- this lane's neighbor row & its 4 PPF scalars -------------------
        int kk  = rt * 16 + lrow;
        int idx = knn_idx[m * KNN + kk];
        float px = points[3 * idx + 0] - qx;
        float py = points[3 * idx + 1] - qy;
        float pz = points[3 * idx + 2] - qz;
        float bx = points_normals[3 * idx + 0];
        float by = points_normals[3 * idx + 1];
        float bz = points_normals[3 * idx + 2];

        float f0 = sqrtf(px * px + py * py + pz * pz) * SCALE_2PI;
        // angle(pij, n1)
        float cx = py * az - pz * ay, cy = pz * ax - px * az, cz = px * ay - py * ax;
        float f1 = atan2f(sqrtf(cx * cx + cy * cy + cz * cz),
                          px * ax + py * ay + pz * az);
        // angle(-pij, n2)
        float ex = (-py) * bz - (-pz) * by;
        float ey = (-pz) * bx - (-px) * bz;
        float ez = (-px) * by - (-py) * bx;
        float f2 = atan2f(sqrtf(ex * ex + ey * ey + ez * ez),
                          -(px * bx + py * by + pz * bz));
        // angle(n1, n2)
        float gx = ay * bz - az * by, gy = az * bx - ax * bz, gz = ax * by - ay * bx;
        float f3 = atan2f(sqrtf(gx * gx + gy * gy + gz * gz),
                          ax * bx + ay * by + az * bz);
        float feat[4] = { f0, f1, f2, f3 };

        v8f acc[4];
#pragma unroll
        for (int nt = 0; nt < 4; ++nt)
#pragma unroll
            for (int v = 0; v < 8; ++v) acc[nt][v] = 0.0f;

#pragma unroll
        for (int kt = 0; kt < 8; ++kt) {        // 8 K-chunks = 4 quarters x 2
            float x = feat[kt >> 1];
            v16h a;
#pragma unroll
            for (int v = 0; v < 8; ++v) {
                int t = (kt & 1) * 16 + v + hi * 8; // 0..31
                float om = exp2f(OMEGA_C_QUART * (float)t);
                float ang = om * x;
                a[2 * v + 0] = (_Float16)__sinf(ang);
                a[2 * v + 1] = (_Float16)__cosf(ang);
            }
#pragma unroll
            for (int nt = 0; nt < 4; ++nt) {
                int ntg = wave * 4 + nt;
                v16h b = smem_frag[(ntg * 8 + kt) * 32 + lane];
                acc[nt] = __builtin_amdgcn_wmma_f32_16x16x32_f16(
                    false, a, false, b, (short)0, acc[nt], false, false);
            }
        }
#pragma unroll
        for (int nt = 0; nt < 4; ++nt)
#pragma unroll
            for (int v = 0; v < 8; ++v)
                vmax[nt][v] = fmaxf(vmax[nt][v], acc[nt][v]);
    }

    // ---- reduce over the 16 rows inside the tile ---------------------------
#pragma unroll
    for (int nt = 0; nt < 4; ++nt) {
        float r = vmax[nt][0];
#pragma unroll
        for (int v = 1; v < 8; ++v) r = fmaxf(r, vmax[nt][v]);
        r = fmaxf(r, __shfl_xor(r, 16, 32));     // M=0..7 vs M=8..15 halves
        if (lane < 16) {
            int col = wave * 64 + nt * 16 + lrow;
            loc[m * D_MODEL + col] = r + b_patch[col];
        }
    }
}

// ---------------------------------------------------------------------------
// Host-side launcher
// ---------------------------------------------------------------------------
extern "C" void kernel_launch(void* const* d_in, const int* in_sizes, int n_in,
                              void* d_out, int out_size, void* d_ws, size_t ws_size,
                              hipStream_t stream) {
    const float* points         = (const float*)d_in[0];
    const float* nodes          = (const float*)d_in[1];
    const float* points_normals = (const float*)d_in[2];
    const float* nodes_normals  = (const float*)d_in[3];
    const float* W_d            = (const float*)d_in[4];
    const float* b_d            = (const float*)d_in[5];
    const float* W_patch        = (const float*)d_in[6];
    const float* b_patch        = (const float*)d_in[7];

    float* glo = (float*)d_out;
    float* loc = glo + (size_t)M_NODES * M_NODES * D_MODEL;

    // workspace carve-up
    char* ws = (char*)d_ws;
    _Float16* wd_frag = (_Float16*)(ws);                       // 128 KB
    _Float16* wp_frag = (_Float16*)(ws + (128 << 10));         // 128 KB
    float*    d2      = (float*)   (ws + (256 << 10));         // 32 MB
    int*      knn_idx = (int*)     (ws + (256 << 10) + (size_t)M_NODES * N_POINTS * 4);

    // 1) swizzle weights into per-lane B-fragment layout (f16)
    prep_wfrag_kernel<<<128, 256, 0, stream>>>(W_d, wd_frag);
    prep_wfrag_kernel<<<128, 256, 0, stream>>>(W_patch, wp_frag);

    // 2) pairwise node->point squared distances
    dim3 gd2(N_POINTS / 256, M_NODES);
    d2_kernel<<<gd2, 256, 0, stream>>>(nodes, points, d2);

    // 3) deterministic top-128 selection per node
    knn_select_kernel<<<M_NODES, 256, 0, stream>>>(d2, knn_idx);

    // 4) glo: (M*M, 256) embedding-GEMM via WMMA, B fragments in LDS
    glo_kernel<<<(M_NODES * M_NODES) / 128, 256, FRAG_BYTES, stream>>>(
        nodes, nodes_normals, wd_frag, b_d, glo);

    // 5) loc: (M*K, 256) embedding-GEMM + segmented max via WMMA
    loc_kernel<<<M_NODES, 128, FRAG_BYTES, stream>>>(
        points, points_normals, nodes, nodes_normals,
        knn_idx, wp_frag, b_patch, loc);
}
